// GATReceptionPredictor_41558103556525
// MI455X (gfx1250) — matrix operand order repaired and checked by reading, not compiled, not run on hardware
//
#include <hip/hip_runtime.h>
#include <hip/hip_bf16.h>
#include <math.h>

// ---------------------------------------------------------------------------
// GATv2 reception predictor for MI455X (gfx1250, wave32, WMMA).
// Dense GEMMs -> v_wmma_f32_16x16x32_f16 with async global->LDS A staging.
// Edge softmax/aggregation -> edge-parallel kernels with float atomics.
// ---------------------------------------------------------------------------

typedef __attribute__((ext_vector_type(16))) _Float16 v16h;
typedef __attribute__((ext_vector_type(8)))  _Float16 v8h;
typedef __attribute__((ext_vector_type(8)))  float    v8f;

#define NN   50000
#define EE   800000
#define HID  128
#define EH   32
#define HEADS 16
#define CD   8

// -------------------- helpers --------------------
__device__ __forceinline__ unsigned fenc(float f) {
    unsigned u = __float_as_uint(f);
    return (u & 0x80000000u) ? ~u : (u | 0x80000000u);
}
__device__ __forceinline__ float fdec(unsigned k) {
    return (k & 0x80000000u) ? __uint_as_float(k & 0x7FFFFFFFu)
                             : __uint_as_float(~k);
}

// build 16-half A/B fragment from LDS f32 row (ISA 7.12.2 16-bit A 16x32):
// halves j<8 -> k=kb+j ; j>=8 -> k=kb+16+(j-8)
__device__ __forceinline__ v16h frag_from_f32(const float* p, int kb) {
    v8f f0 = *(const v8f*)(p + kb);        // k = kb .. kb+7
    v8f f1 = *(const v8f*)(p + kb + 16);   // k = kb+16 .. kb+23
    v8h h0 = __builtin_convertvector(f0, v8h);
    v8h h1 = __builtin_convertvector(f1, v8h);
    return __builtin_shufflevector(h0, h1, 0,1,2,3,4,5,6,7,
                                           8,9,10,11,12,13,14,15);
}
__device__ __forceinline__ v16h frag_from_f16(const _Float16* p) {
    v8h h0 = *(const v8h*)p;
    v8h h1 = *(const v8h*)(p + 16);
    return __builtin_shufflevector(h0, h1, 0,1,2,3,4,5,6,7,
                                           8,9,10,11,12,13,14,15);
}

// -------------------- f32 -> f16 weight conversion --------------------
__global__ void cvt_f16_kernel(const float* __restrict__ src,
                               _Float16* __restrict__ dst, int n) {
    for (int i = blockIdx.x * blockDim.x + threadIdx.x; i < n;
         i += gridDim.x * blockDim.x)
        dst[i] = (_Float16)src[i];
}

// -------------------- WMMA GEMM: C[M,N] = act(A[M,K] @ W[K,N] + bias (+C)) --
// block = 128 threads = 4 waves; block tile = 64 rows x 32 cols.
// Wave tile = 16 rows x 32 cols (one A fragment, two B fragments, 2 WMMAs).
// A tile staged f32 via GLOBAL_LOAD_ASYNC_TO_LDS_B128 (ASYNCcnt), converted
// to f16 per-fragment. All staging is branch-free (clamped addresses).
__global__ void gemm_wmma_f16(const float* __restrict__ A,
                              const _Float16* __restrict__ W,
                              const float* __restrict__ bias,
                              float* __restrict__ C,
                              int M, int K, int N, int act, int accum) {
    __shared__ __align__(32) float    lAf[64][40];   // 64 rows x 32 k (f32, pad)
    __shared__ __align__(16) _Float16 lBt[32][40];   // 32 cols x 32 k (f16, pad)

    const int tid  = threadIdx.x;
    const int wave = tid >> 5;
    const int lane = tid & 31;
    const int rowBase = blockIdx.x * 64;
    const int colBase = blockIdx.y * 32;
    const int hi = (lane >= 16);
    const int mn = lane & 15;   // row (A frag) / col (B,C,D frag) within tile
    const size_t maxFlat = (size_t)M * K - 4;

    v8f acc0 = {}, acc1 = {};
    if (accum) {
#pragma unroll
        for (int r = 0; r < 8; ++r) {
            int gr = rowBase + wave * 16 + r + (hi ? 8 : 0);
            if (gr < M) {
                acc0[r] = C[(size_t)gr * N + colBase + mn];
                acc1[r] = C[(size_t)gr * N + colBase + 16 + mn];
            }
        }
    }

    for (int k0 = 0; k0 < K; k0 += 32) {
        // ---- stage A tile (f32) via async global->LDS, 16B per lane x4 ----
#pragma unroll
        for (int it = 0; it < 4; ++it) {
            int i  = tid + it * 128;          // float4 id: 0..511
            int r  = i >> 3;                  // 8 float4 per 32-wide row
            int c4 = (i & 7) * 4;
            size_t flat = (size_t)(rowBase + r) * K + (size_t)(k0 + c4);
            if (flat > maxFlat) flat = maxFlat;   // clamp: garbage k>=K hits B=0
            const float* gp = A + flat;
            unsigned lds = (unsigned)(uintptr_t)(&lAf[r][c4]);
            asm volatile("global_load_async_to_lds_b128 %0, %1, off"
                         :: "v"(lds), "v"(gp) : "memory");
        }
        // ---- stage B tile transposed (coalesced: lanes sweep N) ----
#pragma unroll
        for (int it = 0; it < 8; ++it) {
            int i  = tid + it * 128;          // 0..1023
            int n2 = i & 31, kk = i >> 5;
            int gk  = k0 + kk;
            int gka = gk < K ? gk : K - 1;
            _Float16 v = W[(size_t)gka * N + colBase + n2];
            lBt[n2][kk] = (gk < K) ? v : (_Float16)0.f;
        }
        asm volatile("s_wait_asynccnt 0x0" ::: "memory");
        __syncthreads();

        const int kbA = hi ? 8 : 0;
        v16h af = frag_from_f32(&lAf[wave * 16 + mn][0], kbA);
        const int kbB = hi ? 8 : 0;
        v16h bf0 = frag_from_f16(&lBt[mn][kbB]);
        v16h bf1 = frag_from_f16(&lBt[16 + mn][kbB]);

        acc0 = __builtin_amdgcn_wmma_f32_16x16x32_f16(
            false, af, false, bf0, (short)0, acc0, false, false);
        acc1 = __builtin_amdgcn_wmma_f32_16x16x32_f16(
            false, af, false, bf1, (short)0, acc1, false, false);
        __syncthreads();
    }

    float bv0 = bias ? bias[colBase + mn] : 0.f;
    float bv1 = bias ? bias[colBase + 16 + mn] : 0.f;
#pragma unroll
    for (int r = 0; r < 8; ++r) {
        int gr = rowBase + wave * 16 + r + (hi ? 8 : 0);
        if (gr < M) {
            float v0 = acc0[r] + bv0;
            float v1 = acc1[r] + bv1;
            if (act == 1) { v0 = v0 > 0.f ? v0 : 0.f; v1 = v1 > 0.f ? v1 : 0.f; }
            C[(size_t)gr * N + colBase + mn]      = v0;
            C[(size_t)gr * N + colBase + 16 + mn] = v1;
        }
    }
}

// -------------------- edge mean features (PyG fill_value='mean') -----------
__global__ void edge_mean_accum(const int* __restrict__ ei,
                                const float* __restrict__ ef,
                                float* __restrict__ cnt,
                                float* __restrict__ macc) {
    int e = blockIdx.x * blockDim.x + threadIdx.x;
    if (e >= EE) return;
    int d = ei[EE + e];
    atomicAdd(&cnt[d], 1.f);
    const float* p = ef + (size_t)e * EH;
    float* q = macc + (size_t)d * EH;
#pragma unroll
    for (int k = 0; k < EH; ++k) atomicAdd(&q[k], p[k]);
}

__global__ void mean_finalize(const float* __restrict__ macc,
                              const float* __restrict__ cnt,
                              float* __restrict__ mean_ef) {
    int i = blockIdx.x * blockDim.x + threadIdx.x;
    if (i >= NN * EH) return;
    mean_ef[i] = macc[i] / fmaxf(cnt[i >> 5], 1.f);
}

// -------------------- GATv2 attention: score + segment max -----------------
// one thread per (edge, head); ee = ef_full @ We fused (avoids 435 MB buffer)
__global__ void gat_score(const int* __restrict__ ei,
                          const float* __restrict__ ef,
                          const float* __restrict__ mean_ef,
                          const float* __restrict__ xl,
                          const float* __restrict__ xr,
                          const float* __restrict__ We,
                          const float* __restrict__ att,
                          float* __restrict__ score,
                          unsigned* __restrict__ mkey) {
    int t = blockIdx.x * blockDim.x + threadIdx.x;
    if (t >= (EE + NN) * HEADS) return;
    int e = t >> 4, hh = t & 15;
    int s, d;
    const float* effp;
    if (e < EE) { s = ei[e]; d = ei[EE + e]; effp = ef + (size_t)e * EH; }
    else        { s = e - EE; d = s;         effp = mean_ef + (size_t)s * EH; }

    float eff[EH];
#pragma unroll
    for (int k = 0; k < EH; ++k) eff[k] = effp[k];

    const float* xlp = xl + (size_t)s * HID + hh * CD;
    const float* xrp = xr + (size_t)d * HID + hh * CD;
    __builtin_prefetch(xlp, 0, 0);

    float sc = 0.f;
#pragma unroll
    for (int c = 0; c < CD; ++c) {
        float eev = 0.f;
#pragma unroll
        for (int k = 0; k < EH; ++k) eev += eff[k] * We[k * HID + hh * CD + c];
        float sv = xlp[c] + xrp[c] + eev;
        sv = sv > 0.f ? sv : 0.2f * sv;          // leaky_relu(0.2)
        sc += sv * att[hh * CD + c];
    }
    score[t] = sc;
    atomicMax(&mkey[d * HEADS + hh], fenc(sc));
}

// -------------------- exp + segment sum ------------------------------------
__global__ void gat_exp(const int* __restrict__ ei,
                        float* __restrict__ score,
                        const unsigned* __restrict__ mkey,
                        float* __restrict__ den) {
    int t = blockIdx.x * blockDim.x + threadIdx.x;
    if (t >= (EE + NN) * HEADS) return;
    int e = t >> 4, hh = t & 15;
    int d = (e < EE) ? ei[EE + e] : (e - EE);
    float ex = __expf(score[t] - fdec(mkey[d * HEADS + hh]));
    score[t] = ex;
    atomicAdd(&den[d * HEADS + hh], ex);
}

// -------------------- weighted aggregation ---------------------------------
__global__ void gat_agg(const int* __restrict__ ei,
                        const float* __restrict__ score,
                        const float* __restrict__ den,
                        const float* __restrict__ xl,
                        float* __restrict__ outacc) {
    int t = blockIdx.x * blockDim.x + threadIdx.x;
    if (t >= (EE + NN) * HEADS) return;
    int e = t >> 4, hh = t & 15;
    int s, d;
    if (e < EE) { s = ei[e]; d = ei[EE + e]; }
    else        { s = e - EE; d = s; }
    float alpha = score[t] / den[d * HEADS + hh];
    const float* xlp = xl + (size_t)s * HID + hh * CD;
    float* op = outacc + (size_t)d * HID + hh * CD;
#pragma unroll
    for (int c = 0; c < CD; ++c) atomicAdd(&op[c], alpha * xlp[c]);
}

// -------------------- bias + relu (in place) -------------------------------
__global__ void bias_relu(float* __restrict__ h, const float* __restrict__ b) {
    int i = blockIdx.x * blockDim.x + threadIdx.x;
    if (i >= NN * HID) return;
    float v = h[i] + b[i & (HID - 1)];
    h[i] = v > 0.f ? v : 0.f;
}

// -------------------- MLP head GEMV + sigmoid ------------------------------
__global__ void head_kernel(const float* __restrict__ z,
                            const float* __restrict__ Wm2,
                            const float* __restrict__ bm2,
                            float* __restrict__ out) {
    int i = blockIdx.x * blockDim.x + threadIdx.x;
    if (i >= NN) return;
    const float* zp = z + (size_t)i * HID;
    float t = bm2[0];
#pragma unroll
    for (int k = 0; k < HID; ++k) t += zp[k] * Wm2[k];
    out[i] = 1.f / (1.f + __expf(-t));
}

// ---------------------------------------------------------------------------
extern "C" void kernel_launch(void* const* d_in, const int* in_sizes, int n_in,
                              void* d_out, int out_size, void* d_ws, size_t ws_size,
                              hipStream_t stream) {
    const float* x      = (const float*)d_in[0];
    const int*   ei     = (const int*)d_in[1];
    const float* ea     = (const float*)d_in[2];
    const float* Wn     = (const float*)d_in[4];
    const float* bn     = (const float*)d_in[5];
    const float* We_enc = (const float*)d_in[6];
    const float* be_enc = (const float*)d_in[7];
    const float* Wl1 = (const float*)d_in[8];  const float* bl1 = (const float*)d_in[9];
    const float* Wr1 = (const float*)d_in[10]; const float* br1 = (const float*)d_in[11];
    const float* We1 = (const float*)d_in[12]; const float* att1 = (const float*)d_in[13];
    const float* bias1 = (const float*)d_in[14];
    const float* Wl2 = (const float*)d_in[15]; const float* bl2 = (const float*)d_in[16];
    const float* Wr2 = (const float*)d_in[17]; const float* br2 = (const float*)d_in[18];
    const float* We2 = (const float*)d_in[19]; const float* att2 = (const float*)d_in[20];
    const float* bias2 = (const float*)d_in[21];
    const float* Wm1 = (const float*)d_in[22]; const float* bm1 = (const float*)d_in[23];
    const float* Wm2 = (const float*)d_in[24]; const float* bm2 = (const float*)d_in[25];

    // ---- workspace layout ----
    char* ws = (char*)d_ws;
    size_t off = 0;
    auto carve = [&](size_t bytes) -> void* {
        void* p = ws + off;
        off += (bytes + 255) & ~(size_t)255;
        return p;
    };
    float*    h0      = (float*)carve((size_t)NN * HID * 4);
    float*    h1      = (float*)carve((size_t)NN * HID * 4);
    float*    h2      = (float*)carve((size_t)NN * HID * 4);
    float*    ef      = (float*)carve((size_t)EE * EH * 4);
    float*    macc    = (float*)carve((size_t)NN * EH * 4);
    float*    mean_ef = (float*)carve((size_t)NN * EH * 4);
    float*    cnt     = (float*)carve((size_t)NN * 4);
    float*    xl      = (float*)carve((size_t)NN * HID * 4);
    float*    xr      = (float*)carve((size_t)NN * HID * 4);
    float*    score   = (float*)carve((size_t)(EE + NN) * HEADS * 4);
    unsigned* mkey    = (unsigned*)carve((size_t)NN * HEADS * 4);
    float*    den     = (float*)carve((size_t)NN * HEADS * 4);
    float*    z       = (float*)carve((size_t)NN * HID * 4);
    _Float16* Wn16    = (_Float16*)carve(64 * 128 * 2);
    _Float16* Wenc16  = (_Float16*)carve(16 * 32 * 2);
    _Float16* Wl116   = (_Float16*)carve(128 * 128 * 2);
    _Float16* Wr116   = (_Float16*)carve(128 * 128 * 2);
    _Float16* Wl216   = (_Float16*)carve(128 * 128 * 2);
    _Float16* Wr216   = (_Float16*)carve(128 * 128 * 2);
    _Float16* Wm116   = (_Float16*)carve(256 * 128 * 2);

    // ---- convert weights to f16 ----
    auto cvt = [&](const float* s, _Float16* d, int n) {
        cvt_f16_kernel<<<(n + 255) / 256, 256, 0, stream>>>(s, d, n);
    };
    cvt(Wn, Wn16, 64 * 128);
    cvt(We_enc, Wenc16, 16 * 32);
    cvt(Wl1, Wl116, 128 * 128);
    cvt(Wr1, Wr116, 128 * 128);
    cvt(Wl2, Wl216, 128 * 128);
    cvt(Wr2, Wr216, 128 * 128);
    cvt(Wm1, Wm116, 256 * 128);

    dim3 blk(128);
    auto gemm = [&](const float* A, const _Float16* W, const float* b,
                    float* C, int M, int K, int N_, int act, int accum) {
        dim3 grid((M + 63) / 64, N_ / 32);
        gemm_wmma_f16<<<grid, blk, 0, stream>>>(A, W, b, C, M, K, N_, act, accum);
    };

    // ---- encoders ----
    gemm(x, Wn16, bn, h0, NN, 64, HID, 0, 0);       // h0 = x@Wn + bn
    gemm(ea, Wenc16, be_enc, ef, EE, 16, EH, 0, 0); // ef = ea@We_enc + be_enc

    // ---- self-loop mean edge features ----
    hipMemsetAsync(cnt, 0, (size_t)NN * 4, stream);
    hipMemsetAsync(macc, 0, (size_t)NN * EH * 4, stream);
    edge_mean_accum<<<(EE + 255) / 256, 256, 0, stream>>>(ei, ef, cnt, macc);
    mean_finalize<<<(NN * EH + 255) / 256, 256, 0, stream>>>(macc, cnt, mean_ef);

    const int etot = (EE + NN) * HEADS;
    auto layer = [&](const float* hin, const _Float16* WlX, const float* blX,
                     const _Float16* WrX, const float* brX,
                     const float* WeX, const float* attX, const float* biasX,
                     float* hout) {
        gemm(hin, WlX, blX, xl, NN, HID, HID, 0, 0);
        gemm(hin, WrX, brX, xr, NN, HID, HID, 0, 0);
        hipMemsetAsync(mkey, 0, (size_t)NN * HEADS * 4, stream); // -inf sentinel
        hipMemsetAsync(den, 0, (size_t)NN * HEADS * 4, stream);
        hipMemsetAsync(hout, 0, (size_t)NN * HID * 4, stream);
        gat_score<<<(etot + 255) / 256, 256, 0, stream>>>(ei, ef, mean_ef, xl, xr,
                                                          WeX, attX, score, mkey);
        gat_exp<<<(etot + 255) / 256, 256, 0, stream>>>(ei, score, mkey, den);
        gat_agg<<<(etot + 255) / 256, 256, 0, stream>>>(ei, score, den, xl, hout);
        bias_relu<<<(NN * HID + 255) / 256, 256, 0, stream>>>(hout, biasX);
    };

    layer(h0, Wl116, bl1, Wr116, br1, We1, att1, bias1, h1);
    layer(h1, Wl216, bl2, Wr216, br2, We2, att2, bias2, h2);

    // ---- MLP head: z = relu([h0,h2]@Wm1 + bm1), fused as two GEMM passes ----
    gemm(h0, Wm116, nullptr, z, NN, HID, HID, 0, 0);            // z  = h0@Wm1_top
    gemm(h2, Wm116 + 128 * 128, bm1, z, NN, HID, HID, 1, 1);    // z += h2@Wm1_bot; relu
    head_kernel<<<(NN + 255) / 256, 256, 0, stream>>>(z, Wm2, bm2, (float*)d_out);
}
